// MultiHeadAttention_60739427500331
// MI455X (gfx1250) — matrix operand order repaired and checked by reading, not compiled, not run on hardware
//
#include <hip/hip_runtime.h>

#define D_MODEL 1024
#define NUM_HEADS 16
#define DEPTH 64
#define BATCH 4
#define SEQ 2048
#define MTOT (BATCH * SEQ)   // 8192

typedef __attribute__((ext_vector_type(16))) _Float16 v16h;
typedef __attribute__((ext_vector_type(8)))  _Float16 v8h;
typedef __attribute__((ext_vector_type(8)))  float    v8f;
typedef __attribute__((ext_vector_type(4)))  unsigned int u32x4;
typedef __attribute__((ext_vector_type(4)))  int      i32x4;
typedef __attribute__((ext_vector_type(8)))  int      i32x8;

union F16Frag { v16h v; v8h h[2]; };

__device__ __forceinline__ v8f zero_v8f() {
  v8f z;
#pragma unroll
  for (int i = 0; i < 8; ++i) z[i] = 0.0f;
  return z;
}

// A-fragment (16x32 f16, MxK): lane holds row m=lane&15.
// h[0..7] = row[lo .. lo+7], h[8..15] = row[16+lo .. 16+lo+7], lo=(lane>=16)?8:0
__device__ __forceinline__ v16h load_afrag(const _Float16* __restrict__ row, int lo) {
  F16Frag f;
  f.h[0] = *(const v8h*)(row + lo);
  f.h[1] = *(const v8h*)(row + 16 + lo);
  return f.v;
}

// B-fragment (32x16 f16, KxN): lane holds column n=lane&15; same k order as A:
// h[0..7] = k[lo..lo+7], h[8..15] = k[16+lo..], `col` = base + 16*(lane-half).
__device__ __forceinline__ v16h load_bfrag(const _Float16* __restrict__ col) {
  F16Frag f;
  const _Float16* p = col - ((threadIdx.x & 16) ? 8 : 0);
  f.h[0] = *(const v8h*)(p);
  f.h[1] = *(const v8h*)(p + 16);
  return f.v;
}

__device__ __forceinline__ v8f wmma16(v16h a, v16h b, v8f c) {
  return __builtin_amdgcn_wmma_f32_16x16x32_f16(false, a, false, b, (short)0, c,
                                                false, false);
}

// ---- CDNA5 async copy: per-lane 16B global -> LDS, tracked by ASYNCcnt ----
__device__ __forceinline__ void async_load_b128(const void* gsrc, void* ldsdst) {
  unsigned int l = (unsigned int)(size_t)ldsdst;  // flat[31:0] == LDS offset
  asm volatile("global_load_async_to_lds_b128 %0, %1, off"
               :: "v"(l), "v"(gsrc)
               : "memory");
}
__device__ __forceinline__ void wait_async0() {
  asm volatile("s_wait_asynccnt 0x0" ::: "memory");
}

#if __has_builtin(__builtin_amdgcn_tensor_load_to_lds)
#define HAVE_TDM 1
// TDM load: D# groups built per CDNA5 ISA
__device__ __forceinline__ void tdm_load(u32x4 g0, i32x8 g1) {
  i32x4 z4 = {0, 0, 0, 0};
#if defined(__clang_major__) && __clang_major__ >= 23
  i32x8 z8 = {0, 0, 0, 0, 0, 0, 0, 0};
  __builtin_amdgcn_tensor_load_to_lds(g0, g1, z4, z4, z8, 0);
#else
  __builtin_amdgcn_tensor_load_to_lds(g0, g1, z4, z4, 0);
#endif
}
#else
#define HAVE_TDM 0
#endif

// ---------------- conversion kernels ----------------

__global__ void cvt_f32_to_f16(const float* __restrict__ in,
                               _Float16* __restrict__ out, int n) {
  int i = blockIdx.x * blockDim.x + threadIdx.x;
  if (i < n) out[i] = (_Float16)in[i];
}

// w [K=1024][N=1024] f32 -> wT [N][K] f16
__global__ void cvt_transpose_f16(const float* __restrict__ w,
                                  _Float16* __restrict__ wT) {
  int i = blockIdx.x * blockDim.x + threadIdx.x;
  int k = i >> 10;
  int n = i & 1023;
  wT[(size_t)n * 1024 + k] = (_Float16)w[i];
}

// ---------------- WMMA GEMM: C = A[M x 1024] * BT^T + bias ----------------
// BT is [Ntot x 1024] f16 (pre-transposed weight, row n contiguous over k).
// Block tile 256(M) x 64(N); 8 waves, each wave 32 rows (8 WMMA / K-step).
// mode 0: f16 [m][n]   mode 1: f16 [b][h][s][d]   mode 2: f16 [b][h][d][s]
// mode 3: f32 [m][n]
__global__ __launch_bounds__(256) void gemm_wmma(
    const _Float16* __restrict__ A, const _Float16* __restrict__ BT,
    const float* __restrict__ bias, void* __restrict__ Cout,
    int mode, float scale) {
  __shared__ alignas(16) _Float16 Bs[64 * 32];  // 64 n-rows x 32 k, 4 KB

  const int lane = threadIdx.x & 31;
  const int wv   = threadIdx.x >> 5;            // 0..7
  const int mbase = blockIdx.y * 256 + wv * 32;
  const int nbase = blockIdx.x * 64;
  const int lo  = (lane & 16) ? 8 : 0;
  const int bko = (lane & 16) ? 16 : 0;
  const int mro = (lane & 16) ? 8 : 0;

  const _Float16* arow0 = A + (size_t)(mbase + (lane & 15)) * 1024;
  const _Float16* arow1 = arow0 + (size_t)16 * 1024;

  v8f acc[2][4];
#pragma unroll
  for (int mi = 0; mi < 2; ++mi)
#pragma unroll
    for (int nt = 0; nt < 4; ++nt) acc[mi][nt] = zero_v8f();

#if !HAVE_TDM
  const int ldr = threadIdx.x >> 2;        // 0..63
  const int ldc = (threadIdx.x & 3) * 8;   // 0,8,16,24
  const _Float16* bsrc = BT + (size_t)(nbase + ldr) * 1024 + ldc;
#else
  const unsigned int bs_lds = (unsigned int)(size_t)&Bs[0];
#endif

  for (int kb = 0; kb < 1024; kb += 32) {
    __syncthreads();
#if HAVE_TDM
    if (wv == 0) {
      // D# group0: count=1 | lds_addr | global_addr(57b) | type=2
      unsigned long long ga =
          (unsigned long long)(size_t)(BT + (size_t)nbase * 1024 + kb);
      u32x4 g0;
      g0[0] = 1u;
      g0[1] = bs_lds;
      g0[2] = (unsigned int)ga;
      g0[3] = (unsigned int)(ga >> 32) | (2u << 30);
      // D# group1: data_size=2B; tensor 1024x1024; tile 32(k) x 64(n);
      // tensor_dim0_stride=1024 elements
      i32x8 g1;
      g1[0] = (int)(1u << 16);            // data_size=1 -> 2 bytes
      g1[1] = (int)(1024u << 16);         // tensor_dim0[15:0]
      g1[2] = (int)(1024u << 16);         // tensor_dim0[31:16]=0 | tensor_dim1[15:0]
      g1[3] = (int)(32u << 16);           // tensor_dim1[31:16]=0 | tile_dim0=32
      g1[4] = 64;                         // tile_dim1=64 | tile_dim2=0
      g1[5] = 1024;                       // tensor_dim0_stride[31:0]
      g1[6] = 0;                          // stride hi | tensor_dim1_stride lo
      g1[7] = 0;
      tdm_load(g0, g1);
      __builtin_amdgcn_s_wait_tensorcnt(0);
    }
#else
    *(v8h*)&Bs[ldr * 32 + ldc] = *(const v8h*)(bsrc + kb);
#endif
    __syncthreads();

    v16h af0 = load_afrag(arow0 + kb, lo);
    v16h af1 = load_afrag(arow1 + kb, lo);
#pragma unroll
    for (int nt = 0; nt < 4; ++nt) {
      v16h bf = load_bfrag(&Bs[(nt * 16 + (lane & 15)) * 32 + bko]);
      acc[0][nt] = wmma16(af0, bf, acc[0][nt]);
      acc[1][nt] = wmma16(af1, bf, acc[1][nt]);
    }
  }

#pragma unroll
  for (int nt = 0; nt < 4; ++nt) {
    const int n = nbase + nt * 16 + (lane & 15);
    const float bv = bias[n];
#pragma unroll
    for (int mi = 0; mi < 2; ++mi)
#pragma unroll
      for (int r = 0; r < 8; ++r) {
        const float val = (acc[mi][nt][r] + bv) * scale;
        const int m = mbase + mi * 16 + r + mro;
        if (mode == 3) {
          ((float*)Cout)[(size_t)m * D_MODEL + n] = val;
        } else if (mode == 0) {
          ((_Float16*)Cout)[(size_t)m * D_MODEL + n] = (_Float16)val;
        } else {
          const int b = m >> 11, s = m & (SEQ - 1);
          const int h = n >> 6,  d = n & (DEPTH - 1);
          size_t idx;
          if (mode == 1)
            idx = (((size_t)b * NUM_HEADS + h) * SEQ + s) * DEPTH + d;
          else
            idx = (((size_t)b * NUM_HEADS + h) * DEPTH + d) * SEQ + s;
          ((_Float16*)Cout)[idx] = (_Float16)val;
        }
      }
  }
}

// ---------------- flash attention ----------------
// Qh, Kh: [B][H][S][DEPTH] f16 (Q pre-scaled by 1/sqrt(DEPTH))
// Vt:     [B][H][DEPTH][S] f16
// Oh:     [B][S][D_MODEL]  f16
// 4 waves/block share one (b,h); K/V key-chunks staged once per block in LDS
// via async-to-LDS copies; each wave owns a 16-query tile.
__global__ __launch_bounds__(128) void flash_attn(
    const _Float16* __restrict__ Qh, const _Float16* __restrict__ Kh,
    const _Float16* __restrict__ Vt, _Float16* __restrict__ Oh) {
  __shared__ alignas(16) _Float16 Ks[32 * 64];   // [key][depth]  4 KB
  __shared__ alignas(16) _Float16 Vs[64 * 32];   // [depth][key]  4 KB
  __shared__ float                Sbuf[4][16 * 32];
  __shared__ alignas(16) _Float16 Pbuf[4][16 * 32];
  __shared__ float                Fac[4][16];

  const int t    = threadIdx.x;
  const int lane = t & 31;
  const int wv   = t >> 5;
  const int h = blockIdx.y, b = blockIdx.z;
  const int qbase = (blockIdx.x * 4 + wv) * 16;
  const size_t bh = (size_t)b * NUM_HEADS + h;

  const _Float16* Qp = Qh + (bh * SEQ + qbase) * DEPTH;
  const _Float16* Kp = Kh + bh * SEQ * DEPTH;
  const _Float16* Vp = Vt + bh * (size_t)DEPTH * SEQ;

  const int lo  = (lane & 16) ? 8 : 0;
  const int bko = (lane & 16) ? 16 : 0;
  const int mro = (lane & 16) ? 8 : 0;

  // staging: K tile 32x64 halfs = 256 16B-chunks; V tile 64x32 likewise.
  const int kr0 = t >> 3,         kc0 = (t & 7) * 8;
  const int kr1 = (t + 128) >> 3, kc1 = ((t + 128) & 7) * 8;
  const int vr0 = t >> 2,         vc0 = (t & 3) * 8;
  const int vr1 = (t + 128) >> 2, vc1 = ((t + 128) & 3) * 8;

  const _Float16* qrow = Qp + (size_t)(lane & 15) * DEPTH;
  const v16h aq0 = load_afrag(qrow +  0, lo);
  const v16h aq1 = load_afrag(qrow + 32, lo);

  v8f o[4];
#pragma unroll
  for (int i = 0; i < 4; ++i) o[i] = zero_v8f();
  float rmax = -3.0e38f, rsum = 0.0f;

  for (int c = 0; c < SEQ; c += 32) {
    // ---- stage shared K/V chunk via async-to-LDS ----
    async_load_b128(Kp + (size_t)(c + kr0) * DEPTH + kc0, &Ks[kr0 * 64 + kc0]);
    async_load_b128(Kp + (size_t)(c + kr1) * DEPTH + kc1, &Ks[kr1 * 64 + kc1]);
    async_load_b128(Vp + (size_t)vr0 * SEQ + c + vc0,     &Vs[vr0 * 32 + vc0]);
    async_load_b128(Vp + (size_t)vr1 * SEQ + c + vc1,     &Vs[vr1 * 32 + vc1]);
    wait_async0();
    __syncthreads();

    // ---- scores S[16 q, 32 keys] = Q * K^T (from LDS) ----
    v8f s0 = zero_v8f(), s1 = zero_v8f();
    {
      const _Float16* k0 = &Ks[(size_t)(lane & 15) * 64 + bko];
      s0 = wmma16(aq0, load_bfrag(k0), s0);
      s0 = wmma16(aq1, load_bfrag(k0 + 32), s0);
      const _Float16* k1 = &Ks[(size_t)(16 + (lane & 15)) * 64 + bko];
      s1 = wmma16(aq0, load_bfrag(k1), s1);
      s1 = wmma16(aq1, load_bfrag(k1 + 32), s1);
    }
#pragma unroll
    for (int r = 0; r < 8; ++r) {
      Sbuf[wv][(r + mro) * 32 + (lane & 15)]      = s0[r];
      Sbuf[wv][(r + mro) * 32 + 16 + (lane & 15)] = s1[r];
    }
    __syncthreads();

    // ---- online softmax (lanes 0..15 own one query row each) ----
    if (lane < 16) {
      float srow[32];
      float mx = rmax;
#pragma unroll
      for (int j = 0; j < 32; ++j) {
        srow[j] = Sbuf[wv][lane * 32 + j];
        mx = fmaxf(mx, srow[j]);
      }
      const float corr = __expf(rmax - mx);
      float psum = 0.0f;
#pragma unroll
      for (int j = 0; j < 32; ++j) {
        const float p = __expf(srow[j] - mx);
        psum += p;
        Pbuf[wv][lane * 32 + j] = (_Float16)p;
      }
      rsum = rsum * corr + psum;
      rmax = mx;
      Fac[wv][lane] = corr;
    }
    __syncthreads();

    // ---- rescale running O, then O += P * V (V from LDS) ----
    float f[8];
#pragma unroll
    for (int r = 0; r < 8; ++r) f[r] = Fac[wv][r + mro];
#pragma unroll
    for (int nt = 0; nt < 4; ++nt)
#pragma unroll
      for (int r = 0; r < 8; ++r) o[nt][r] *= f[r];

    const v16h ap = load_afrag(&Pbuf[wv][(lane & 15) * 32], lo);
#pragma unroll
    for (int nt = 0; nt < 4; ++nt) {
      v16h bf = load_bfrag(&Vs[(size_t)(nt * 16 + (lane & 15)) * 32 + bko]);
      o[nt] = wmma16(ap, bf, o[nt]);
    }
    __syncthreads();  // protect Ks/Vs before next chunk overwrites them
  }

  // ---- finalize: divide by row sums, write [b][q][h*64+d] ----
  if (lane < 16) Fac[wv][lane] = 1.0f / rsum;
  __syncthreads();
  float inv[8];
#pragma unroll
  for (int r = 0; r < 8; ++r) inv[r] = Fac[wv][r + mro];

  const int hbase = h * DEPTH;
#pragma unroll
  for (int nt = 0; nt < 4; ++nt) {
    const int d = nt * 16 + (lane & 15);
#pragma unroll
    for (int r = 0; r < 8; ++r) {
      const int q = qbase + r + mro;
      Oh[((size_t)b * SEQ + q) * D_MODEL + hbase + d] = (_Float16)(o[nt][r] * inv[r]);
    }
  }
}

// ---------------- launcher ----------------

extern "C" void kernel_launch(void* const* d_in, const int* in_sizes, int n_in,
                              void* d_out, int out_size, void* d_ws, size_t ws_size,
                              hipStream_t stream) {
  const float* x  = (const float*)d_in[0];
  const float* wq = (const float*)d_in[1];
  const float* bq = (const float*)d_in[2];
  const float* wk = (const float*)d_in[3];
  const float* bk = (const float*)d_in[4];
  const float* wv = (const float*)d_in[5];
  const float* bv = (const float*)d_in[6];
  const float* wo = (const float*)d_in[7];
  const float* bo = (const float*)d_in[8];

  char* ws = (char*)d_ws;
  const size_t XH_B  = (size_t)MTOT * D_MODEL * 2;
  const size_t W_B   = (size_t)D_MODEL * D_MODEL * 2;
  const size_t HSD_B = (size_t)BATCH * NUM_HEADS * SEQ * DEPTH * 2;

  _Float16* xh  = (_Float16*)(ws);
  _Float16* wqT = (_Float16*)(ws + XH_B);
  _Float16* wkT = (_Float16*)(ws + XH_B + W_B);
  _Float16* wvT = (_Float16*)(ws + XH_B + 2 * W_B);
  _Float16* woT = (_Float16*)(ws + XH_B + 3 * W_B);
  _Float16* Qh  = (_Float16*)(ws + XH_B + 4 * W_B);
  _Float16* Kh  = (_Float16*)(ws + XH_B + 4 * W_B + HSD_B);
  _Float16* Vt  = (_Float16*)(ws + XH_B + 4 * W_B + 2 * HSD_B);
  _Float16* Oh  = (_Float16*)(ws + XH_B + 4 * W_B + 3 * HSD_B);

  {
    const int n = MTOT * D_MODEL;
    cvt_f32_to_f16<<<n / 256, 256, 0, stream>>>(x, xh, n);
    const int nw = D_MODEL * D_MODEL;
    cvt_transpose_f16<<<nw / 256, 256, 0, stream>>>(wq, wqT);
    cvt_transpose_f16<<<nw / 256, 256, 0, stream>>>(wk, wkT);
    cvt_transpose_f16<<<nw / 256, 256, 0, stream>>>(wv, wvT);
    cvt_transpose_f16<<<nw / 256, 256, 0, stream>>>(wo, woT);
  }

  dim3 ggrid(D_MODEL / 64, MTOT / 256);  // (16, 32)
  gemm_wmma<<<ggrid, 256, 0, stream>>>(xh, wqT, bq, (void*)Qh, 1, 0.125f);
  gemm_wmma<<<ggrid, 256, 0, stream>>>(xh, wkT, bk, (void*)Kh, 1, 1.0f);
  gemm_wmma<<<ggrid, 256, 0, stream>>>(xh, wvT, bv, (void*)Vt, 2, 1.0f);

  dim3 agrid(SEQ / 64, NUM_HEADS, BATCH);  // (32, 16, 4)
  flash_attn<<<agrid, 128, 0, stream>>>(Qh, Kh, Vt, Oh);

  gemm_wmma<<<ggrid, 256, 0, stream>>>(Oh, woT, bo, d_out, 3, 1.0f);
}
